// BiteNet_80496277061845
// MI455X (gfx1250) — compile-verified, matrix operand order checked by reading, not run on hardware
//
// BiteNet forward for MI455X (gfx1250), CDNA5 WMMA path, round 2.
//
// ~100 GFLOP of GEMM vs ~0.5 GB traffic -> matrix pipes. All linears use
// v_wmma_f32_16x16x32_bf16 with 16x64 tiles per wave (A fragment reused x4).
// Every WMMA input has a bf16 shadow written by its producer, so both operands
// are straight b128 loads (no per-k cvt VALU). LayerNorm/softmax/residual/mask
// math stays fp32. Epilogues are compile-time templates (branch-free stores).
#include <hip/hip_runtime.h>

typedef unsigned short u16;
typedef __attribute__((ext_vector_type(16))) __bf16 v16bf;
typedef __attribute__((ext_vector_type(8)))  float  v8f;

union BF16V {
  v16bf v;
  u16   u[16];
  uint4 q[2];
};

__device__ __forceinline__ u16 f2bf(float f) {
  unsigned x = __builtin_bit_cast(unsigned int, f);
  return (u16)((x + 0x7fffu + ((x >> 16) & 1u)) >> 16);  // RNE f32->bf16
}

// 16x32 bf16 fragment from row-major bf16 (A operand, or B via B^T-row-major).
// ISA 7.12.2 lane layout:
//  lane<16:  row=lane,    elems 0..7 -> k=0..7,  elems 8..15 -> k=16..23
//  lane>=16: row=lane-16, elems 0..7 -> k=8..15, elems 8..15 -> k=24..31
__device__ __forceinline__ v16bf load_frag(const u16* __restrict__ p0, int ld) {
  const int lane = threadIdx.x & 31;
  const u16* p = p0 + (size_t)(lane & 15) * ld + ((lane >> 4) << 3);
  BF16V a;
  a.q[0] = *reinterpret_cast<const uint4*>(p);
  a.q[1] = *reinterpret_cast<const uint4*>(p + 16);
  return a.v;
}

// B-operand gathered from a column slice of row-major bf16 V[T,256] (for P@V),
// with k bounds guard for the T=48 -> 64 padding case.
__device__ __forceinline__ v16bf load_frag_col(const u16* __restrict__ p0, int ld, int krem) {
  const int lane = threadIdx.x & 31;
  const int n = lane & 15, ko = (lane >> 4) << 3;
  BF16V a;
#pragma unroll
  for (int i = 0; i < 8; i++) {
    int k0 = ko + i, k1 = ko + 16 + i;
    a.u[i]     = (k0 < krem) ? p0[(size_t)k0 * ld + n] : (u16)0;
    a.u[i + 8] = (k1 < krem) ? p0[(size_t)k1 * ld + n] : (u16)0;
  }
  return a.v;
}

// ---------------------------------------------------------------------------
// GEMM: C[M,N] = ep(A[M,K] @ Wt[N,K]^T + bias). 4 waves/block, 16x64 per wave.
// EP: 0 none, 1 relu, 2 sigmoid, 3 multiply by rowmask[row].
// OM: 0 f32 out, 1 bf16 out, 2 both.
template <int EP, int OM>
__global__ void gemm_wmma(const u16* __restrict__ A, const u16* __restrict__ Wt,
                          const float* __restrict__ bias,
                          float* __restrict__ Cf, u16* __restrict__ Cb,
                          int M, int N, int K, const float* __restrict__ rowmask) {
  const int wave = threadIdx.x >> 5;
  const int m0 = (blockIdx.x * 4 + wave) * 16;
  if (m0 >= M) return;
  const int n0 = blockIdx.y * 64;
  const int lane = threadIdx.x & 31;
  v8f acc[4] = {{}, {}, {}, {}};
  for (int k0 = 0; k0 < K; k0 += 32) {
    v16bf a = load_frag(A + (size_t)m0 * K + k0, K);
#pragma unroll
    for (int j = 0; j < 4; j++) {
      v16bf b = load_frag(Wt + (size_t)(n0 + 16 * j) * K + k0, K);
      acc[j] = __builtin_amdgcn_wmma_f32_16x16x32_bf16(false, a, false, b, (short)0, acc[j], false, false);
    }
  }
  const int cl = lane & 15;
  const int rb = (lane < 16) ? 0 : 8;  // C layout: vgpr r -> row rb+r, col = cl
  float rm[8];
  if (EP == 3) {
#pragma unroll
    for (int r = 0; r < 8; r++) rm[r] = rowmask[m0 + rb + r];
  }
#pragma unroll
  for (int j = 0; j < 4; j++) {
    const int n = n0 + 16 * j + cl;
    const float bn = bias ? bias[n] : 0.f;
#pragma unroll
    for (int r = 0; r < 8; r++) {
      const int row = m0 + rb + r;
      float v = acc[j][r] + bn;
      if (EP == 1) v = fmaxf(v, 0.f);
      if (EP == 2) v = 1.f / (1.f + __expf(-v));
      if (EP == 3) v *= rm[r];
      const size_t idx = (size_t)row * N + n;
      if (OM == 0 || OM == 2) Cf[idx] = v;
      if (OM == 1 || OM == 2) Cb[idx] = f2bf(v);
    }
  }
}

// ---------------------------------------------------------------------------
// Fused MHA for one (seq, head, 16-query block), all-bf16 operands, fp32
// softmax. Writes y in place over Q (each wave touches only its own Q tile).
// maskType: 0 diag (block q==k), 1 forward (block q>k), 2 backward (block q<k)
template <int T>
__global__ void attn_kernel(u16* __restrict__ Qy, const u16* __restrict__ Kb,
                            const u16* __restrict__ Vb, int maskType) {
  constexpr int NK = T / 16;
  __shared__ u16 Pl[16 * 64];
  const int lane = threadIdx.x & 31;
  const int s = blockIdx.x;
  const int h = blockIdx.y / NK;
  const int q0 = (blockIdx.y % NK) * 16;
  const size_t base = (size_t)s * T * 256;
  const int cl = lane & 15;
  const int rb = (lane < 16) ? 0 : 8;

  v16bf aq = load_frag(Qy + base + (size_t)q0 * 256 + h * 32, 256);
  v8f sc[NK];
#pragma unroll
  for (int kt = 0; kt < NK; kt++) {
    v16bf bk = load_frag(Kb + base + (size_t)(kt * 16) * 256 + h * 32, 256);
    v8f z = {};
    sc[kt] = __builtin_amdgcn_wmma_f32_16x16x32_bf16(false, aq, false, bk, (short)0, z, false, false);
  }
  const float scale = 0.17677669529663687f;  // 1/sqrt(32)
#pragma unroll
  for (int r = 0; r < 8; r++) {
    const int qrow = q0 + rb + r;
    float mx = -3.4e38f;
#pragma unroll
    for (int kt = 0; kt < NK; kt++) {
      int kcol = kt * 16 + cl;
      float v = sc[kt][r] * scale;
      bool blocked = (maskType == 0) ? (qrow == kcol)
                   : (maskType == 1) ? (qrow > kcol) : (qrow < kcol);
      v = blocked ? -3.0e38f : v;
      sc[kt][r] = v;
      mx = fmaxf(mx, v);
    }
#pragma unroll
    for (int o = 1; o < 16; o <<= 1) mx = fmaxf(mx, __shfl_xor(mx, o, 32));
    float sum = 0.f;
#pragma unroll
    for (int kt = 0; kt < NK; kt++) {
      float e = __expf(sc[kt][r] - mx);
      sc[kt][r] = e;
      sum += e;
    }
#pragma unroll
    for (int o = 1; o < 16; o <<= 1) sum += __shfl_xor(sum, o, 32);
    const float inv = 1.f / sum;
#pragma unroll
    for (int kt = 0; kt < NK; kt++) Pl[(rb + r) * 64 + kt * 16 + cl] = f2bf(sc[kt][r] * inv);
    if (T < 64) Pl[(rb + r) * 64 + 48 + cl] = (u16)0;  // zero-pad K 48..63
  }
  __syncthreads();

  v8f y0 = {}, y1 = {};
#pragma unroll
  for (int kk = 0; kk < 64; kk += 32) {
    v16bf ap = load_frag(Pl + kk, 64);  // P tile as A operand (row-major in LDS)
    const int krem = T - kk;
    v16bf b0 = load_frag_col(Vb + base + (size_t)kk * 256 + h * 32, 256, krem);
    y0 = __builtin_amdgcn_wmma_f32_16x16x32_bf16(false, ap, false, b0, (short)0, y0, false, false);
    v16bf b1 = load_frag_col(Vb + base + (size_t)kk * 256 + h * 32 + 16, 256, krem);
    y1 = __builtin_amdgcn_wmma_f32_16x16x32_bf16(false, ap, false, b1, (short)0, y1, false, false);
  }
  u16* yout = Qy + base + (size_t)q0 * 256 + h * 32;
#pragma unroll
  for (int r = 0; r < 8; r++) {
    yout[(size_t)(rb + r) * 256 + cl]      = f2bf(y0[r]);
    yout[(size_t)(rb + r) * 256 + 16 + cl] = f2bf(y1[r]);
  }
}

// ---------------------------------------------------------------------------
// Out = LN(X)*g + b + R   (fp32), optional bf16 shadow for the next GEMM.
__global__ void ln_residual(const float* __restrict__ X, const float* __restrict__ R,
                            const float* __restrict__ g, const float* __restrict__ b,
                            float* __restrict__ Outf, u16* __restrict__ Outb) {
  __shared__ float red[256];
  const int row = blockIdx.x, d = threadIdx.x;
  const size_t idx = (size_t)row * 256 + d;
  const float v = X[idx];
  red[d] = v; __syncthreads();
  for (int s = 128; s > 0; s >>= 1) { if (d < s) red[d] += red[d + s]; __syncthreads(); }
  const float mu = red[0] * (1.f / 256.f); __syncthreads();
  const float c = v - mu;
  red[d] = c * c; __syncthreads();
  for (int s = 128; s > 0; s >>= 1) { if (d < s) red[d] += red[d + s]; __syncthreads(); }
  const float var = red[0] * (1.f / 256.f);
  const float o = c * rsqrtf(var + 1e-6f) * g[d] + b[d] + R[idx];
  Outf[idx] = o;
  if (Outb) Outb[idx] = f2bf(o);
}

__global__ void gather_emb(const int* __restrict__ x, const float* __restrict__ emb,
                           float* __restrict__ Ef, u16* __restrict__ Eb,
                           float* __restrict__ maskc) {
  const int tok = blockIdx.x, d = threadIdx.x;
  const int idx = x[tok];
  const float v = emb[(size_t)idx * 256 + d];
  Ef[(size_t)tok * 256 + d] = v;
  Eb[(size_t)tok * 256 + d] = f2bf(v);
  if (d == 0) maskc[tok] = (idx != 0) ? 1.f : 0.f;
}

__global__ void maskv_kernel(const float* __restrict__ maskc, float* __restrict__ maskv, int S, int Cc) {
  const int s = blockIdx.x * blockDim.x + threadIdx.x;
  if (s >= S) return;
  float sum = 0.f;
  for (int c = 0; c < Cc; c++) sum += maskc[s * Cc + c];
  maskv[s] = sum;
}

// softmax over axis=1 (T) per channel, weighted sum, optional int_emb add,
// dual f32/bf16 output (either may be null).
__global__ void attn_pool_kernel(const float* __restrict__ U, const float* __restrict__ Hin,
                                 float* __restrict__ Outf, u16* __restrict__ Outb,
                                 int T, int outLd, int outOff,
                                 const float* __restrict__ int_emb, const int* __restrict__ intervals) {
  const int s = blockIdx.x, d = threadIdx.x;
  const size_t base = (size_t)s * T * 256 + d;
  float m = -3.4e38f;
  for (int t = 0; t < T; t++) m = fmaxf(m, U[base + (size_t)t * 256]);
  float sum = 0.f, acc = 0.f;
  for (int t = 0; t < T; t++) {
    float e = __expf(U[base + (size_t)t * 256] - m);
    sum += e;
    acc += e * Hin[base + (size_t)t * 256];
  }
  acc /= sum;
  if (int_emb) acc += int_emb[(size_t)intervals[s] * 256 + d];
  const size_t o = (size_t)s * outLd + outOff + d;
  if (Outf) Outf[o] = acc;
  if (Outb) Outb[o] = f2bf(acc);
}

__global__ void wt_convert(const float* __restrict__ W, u16* __restrict__ Wt, int K, int N) {
  const int i = blockIdx.x * 256 + threadIdx.x;
  if (i >= K * N) return;
  const int k = i / N, n = i - k * N;
  Wt[(size_t)n * K + k] = f2bf(W[i]);  // Wt[N][K] = W[K][N]^T in bf16
}

// ---------------------------------------------------------------------------
struct EncW {
  const u16 *wq, *wk, *wv, *wp, *wf;
  const float *bq, *bk, *bv, *bp, *bf, *g, *b;
};

template <int EP, int OM>
static void gemm(const u16* A, const u16* Wt, const float* bias, float* Cf, u16* Cb,
                 int M, int N, int K, const float* rowmask, hipStream_t st) {
  dim3 grid((M / 16 + 3) / 4, N / 64);
  gemm_wmma<EP, OM><<<grid, 128, 0, st>>>(A, Wt, bias, Cf, Cb, M, N, K, rowmask);
}

// Internal shadows: X1 bf16 -> Kb (dead after attn), H bf16 -> Vb (dead after attn).
// X2f may alias Pf (P is dead after LN1). Hf may alias Ef (E dead after LN1).
static void run_encoder(const u16* Eb, const float* Ef,
                        u16* Qb, u16* Kb, u16* Vb,
                        float* Pf, float* X1f, float* X2f, float* Hf,
                        int S, int T, int maskType, const float* rowmask,
                        const EncW& w, hipStream_t st) {
  const int M = S * T;
  gemm<0, 1>(Eb, w.wq, w.bq, nullptr, Qb, M, 256, 256, nullptr, st);
  gemm<0, 1>(Eb, w.wk, w.bk, nullptr, Kb, M, 256, 256, nullptr, st);
  gemm<0, 1>(Eb, w.wv, w.bv, nullptr, Vb, M, 256, 256, nullptr, st);
  if (T == 48) attn_kernel<48><<<dim3(S, 8 * 3), 32, 0, st>>>(Qb, Kb, Vb, maskType);
  else         attn_kernel<64><<<dim3(S, 8 * 4), 32, 0, st>>>(Qb, Kb, Vb, maskType);
  gemm<3, 0>(Qb, w.wp, w.bp, Pf, nullptr, M, 256, 256, rowmask, st);  // P = proj(y)*mask
  ln_residual<<<M, 256, 0, st>>>(Pf, Ef, w.g, w.b, X1f, Kb);          // X1 = LN(P)+E
  gemm<0, 0>(Kb, w.wf, w.bf, X2f, nullptr, M, 256, 256, nullptr, st); // X2 = fc(X1)
  ln_residual<<<M, 256, 0, st>>>(X2f, X1f, w.g, w.b, Hf, Vb);         // H = LN(X2)+X1
}

static void run_pool(const u16* Hb, const float* Hf, u16* Gb, float* Uf,
                     int S, int T, const float* rowmask,
                     const u16* w1, const float* b1, const u16* w2, const float* b2,
                     float* Outf, u16* Outb, int outLd, int outOff,
                     const float* int_emb, const int* intervals, hipStream_t st) {
  const int M = S * T;
  gemm<1, 1>(Hb, w1, b1, nullptr, Gb, M, 256, 256, nullptr, st);   // G = relu(d1(H))
  gemm<3, 0>(Gb, w2, b2, Uf, nullptr, M, 256, 256, rowmask, st);   // U = d2(G)*mask
  attn_pool_kernel<<<S, 256, 0, st>>>(Uf, Hf, Outf, Outb, T, outLd, outOff, int_emb, intervals);
}

// ---------------------------------------------------------------------------
// Flattened input order (sorted-key pytree flatten of setup_inputs()):
//  0:x  1:intervals  2:emb
//  enc_bwd @3, enc_diag @15, enc_fwd @27; per-encoder (12 leaves):
//    +0 fc.b +1 fc.w +2 k.b +3 k.w +4 ln_b +5 ln_g +6 proj.b +7 proj.w
//    +8 q.b +9 q.w +10 v.b +11 v.w
//  39:fc1.b 40:fc1.w 41:int_emb 42:logits.b 43:logits.w
//  pool1 @44, pool2 @48, pool3 @52: +0 d1.b +1 d1.w +2 d2.b +3 d2.w
extern "C" void kernel_launch(void* const* d_in, const int* in_sizes, int n_in,
                              void* d_out, int out_size, void* d_ws, size_t ws_size,
                              hipStream_t stream) {
  (void)in_sizes; (void)n_in; (void)out_size; (void)ws_size;
  const size_t M1 = 98304, M2 = 2048;  // 32*64*48 tokens, 32*64 visits
  char* ws = (char*)d_ws;
  size_t off = 0;
  auto alloc = [&](size_t bytes) -> char* {
    char* p = ws + off;
    off += (bytes + 255) & ~(size_t)255;
    return p;
  };
  // f32 buffers (code level): F1 = E then H; F2 = P then X2 then U; F3 = X1
  float* F1 = (float*)alloc(M1 * 256 * 4);
  float* F2 = (float*)alloc(M1 * 256 * 4);
  float* F3 = (float*)alloc(M1 * 256 * 4);
  // bf16 shadows (code level)
  u16* Eb = (u16*)alloc(M1 * 256 * 2);
  u16* Qb = (u16*)alloc(M1 * 256 * 2);  // q -> y -> pool G
  u16* Kb = (u16*)alloc(M1 * 256 * 2);  // k -> X1 shadow
  u16* Vb = (u16*)alloc(M1 * 256 * 2);  // v -> H shadow
  // visit level
  float* sEf  = (float*)alloc(M2 * 256 * 4);
  float* sPf  = (float*)alloc(M2 * 256 * 4);  // P / X2 / U
  float* sX1f = (float*)alloc(M2 * 256 * 4);
  float* sHf  = (float*)alloc(M2 * 256 * 4);
  u16* sEb = (u16*)alloc(M2 * 256 * 2);
  u16* sQb = (u16*)alloc(M2 * 256 * 2);
  u16* sKb = (u16*)alloc(M2 * 256 * 2);
  u16* sVb = (u16*)alloc(M2 * 256 * 2);
  float* maskc = (float*)alloc(M1 * 4);
  float* maskv = (float*)alloc(M2 * 4);
  u16* catb = (u16*)alloc(32 * 512 * 2);
  u16* Fb   = (u16*)alloc(32 * 256 * 2);
  auto allocW = [&](size_t elems) -> u16* { return (u16*)alloc(elems * 2); };

  auto F32 = [&](int i) { return (const float*)d_in[i]; };
  auto cvt = [&](int i, u16* Wt, int K, int N) {
    int tot = K * N;
    wt_convert<<<(tot + 255) / 256, 256, 0, stream>>>(F32(i), Wt, K, N);
  };

  // bf16 transposed weights
  u16 *Wq[3], *Wk[3], *Wv[3], *Wp[3], *Wf[3], *Wd1[3], *Wd2[3];
  const int encBase[3] = {3, 15, 27};   // bwd, diag, fwd
  const int poolBase[3] = {44, 48, 52}; // pool1..3
  for (int e = 0; e < 3; e++) {
    Wq[e] = allocW(65536); Wk[e] = allocW(65536); Wv[e] = allocW(65536);
    Wp[e] = allocW(65536); Wf[e] = allocW(65536);
    cvt(encBase[e] + 9,  Wq[e], 256, 256);
    cvt(encBase[e] + 3,  Wk[e], 256, 256);
    cvt(encBase[e] + 11, Wv[e], 256, 256);
    cvt(encBase[e] + 7,  Wp[e], 256, 256);
    cvt(encBase[e] + 1,  Wf[e], 256, 256);
  }
  for (int p = 0; p < 3; p++) {
    Wd1[p] = allocW(65536); Wd2[p] = allocW(65536);
    cvt(poolBase[p] + 1, Wd1[p], 256, 256);
    cvt(poolBase[p] + 3, Wd2[p], 256, 256);
  }
  u16* Wfc1 = allocW(512 * 256); cvt(40, Wfc1, 512, 256);
  u16* Wlg  = allocW(256 * 128); cvt(43, Wlg, 256, 128);

  EncW enc[3];
  for (int e = 0; e < 3; e++) {
    const int b = encBase[e];
    enc[e] = {Wq[e], Wk[e], Wv[e], Wp[e], Wf[e],
              F32(b + 8), F32(b + 2), F32(b + 10), F32(b + 6), F32(b + 0),
              F32(b + 5), F32(b + 4)};
  }

  const int* x = (const int*)d_in[0];
  const int* intervals = (const int*)d_in[1];

  // 1) embedding gather (f32 + bf16) + code mask + visit mask
  gather_emb<<<(int)M1, 256, 0, stream>>>(x, F32(2), F1, Eb, maskc);
  maskv_kernel<<<(int)((M2 + 255) / 256), 256, 0, stream>>>(maskc, maskv, (int)M2, 48);

  // 2) code-level encoder (diag) + pool1 (+ interval embedding) -> sEf/sEb
  run_encoder(Eb, F1, Qb, Kb, Vb, F2, F3, F2, F1, 2048, 48, /*diag*/0, maskc, enc[1], stream);
  run_pool(Vb, F1, Qb, F2, 2048, 48, maskc, Wd1[0], F32(poolBase[0] + 0),
           Wd2[0], F32(poolBase[0] + 2), sEf, sEb, 256, 0, F32(41), intervals, stream);

  // 3) visit-level fwd encoder + pool2 -> cat[:, :256] (bf16)
  run_encoder(sEb, sEf, sQb, sKb, sVb, sPf, sX1f, sPf, sHf, 32, 64, /*fwd*/1, maskv, enc[2], stream);
  run_pool(sVb, sHf, sQb, sPf, 32, 64, maskv, Wd1[1], F32(poolBase[1] + 0),
           Wd2[1], F32(poolBase[1] + 2), nullptr, catb, 512, 0, nullptr, nullptr, stream);

  // 4) visit-level bwd encoder + pool3 -> cat[:, 256:] (bf16); sEf/sEb intact
  run_encoder(sEb, sEf, sQb, sKb, sVb, sPf, sX1f, sPf, sHf, 32, 64, /*bwd*/2, maskv, enc[0], stream);
  run_pool(sVb, sHf, sQb, sPf, 32, 64, maskv, Wd1[2], F32(poolBase[2] + 0),
           Wd2[2], F32(poolBase[2] + 2), nullptr, catb, 512, 256, nullptr, nullptr, stream);

  // 5) head: relu(fc1(cat)) -> sigmoid(logits(.)) -> d_out [32,128] f32
  gemm<1, 1>(catb, Wfc1, F32(39), nullptr, Fb, 32, 256, 512, nullptr, stream);
  gemm<2, 0>(Fb, Wlg, F32(42), (float*)d_out, nullptr, 32, 128, 256, nullptr, stream);
}
